// CausalSelfAttention_32427003084802
// MI455X (gfx1250) — compile-verified, matrix operand order checked by reading, not run on hardware
//
#include <hip/hip_runtime.h>
#include <hip/hip_bf16.h>
#include <math.h>
#include <stdint.h>

typedef __attribute__((ext_vector_type(16))) _Float16 v16h;
typedef __attribute__((ext_vector_type(8)))  _Float16 v8h;
typedef __attribute__((ext_vector_type(8)))  float    v8f;
typedef __attribute__((ext_vector_type(4)))  float    v4f;

#define D_EMBD 768
#define N_HEAD 12
#define HD     64

// ---------------------------------------------------------------------------
// CDNA5 async global->LDS copy (16B per lane), tracked by ASYNCcnt.
// Low 32 bits of a generic __shared__ address == LDS byte offset (flat
// aperture).
// ---------------------------------------------------------------------------
__device__ __forceinline__ void async_ld_b128(uint32_t lds_byte, const void* sbase,
                                              uint32_t vbyte_off) {
    asm volatile("global_load_async_to_lds_b128 %0, %1, %2"
                 :
                 : "v"(lds_byte), "v"(vbyte_off), "s"(sbase)
                 : "memory");
}
__device__ __forceinline__ void wait_async0() {
    asm volatile("s_wait_asynccnt 0x0" ::: "memory");
}
__device__ __forceinline__ void wait_async_le4() {   // prev tile (4 asyncs) done
    asm volatile("s_wait_asynccnt 0x4" ::: "memory");
}

// ---------------------------------------------------------------------------
// Prologue: f32 -> f16 convert (for x), 4 elems/thread.
// ---------------------------------------------------------------------------
__launch_bounds__(256)
__global__ void convert_f16(const float* __restrict__ in, _Float16* __restrict__ out,
                            int n4) {
    int i = (blockIdx.x * 256 + threadIdx.x);
    if (i < n4) {
        v4f v = *(const v4f*)(in + (size_t)i * 4);
        _Float16* o = out + (size_t)i * 4;
        o[0] = (_Float16)v.x; o[1] = (_Float16)v.y;
        o[2] = (_Float16)v.z; o[3] = (_Float16)v.w;
    }
}

// ---------------------------------------------------------------------------
// Prologue: W[k][n] f32  ->  Wt[n][k] f16 (tiled LDS transpose, 32x32 tile).
// ---------------------------------------------------------------------------
__launch_bounds__(256)
__global__ void transpose_convert(const float* __restrict__ W,
                                  _Float16* __restrict__ Wt, int D) {
    __shared__ float t[32][33];
    const int kb = blockIdx.x * 32, nb = blockIdx.y * 32;
    const int tx = threadIdx.x & 31, ty = threadIdx.x >> 5;  // 32 x 8
    #pragma unroll
    for (int r = 0; r < 32; r += 8)
        t[ty + r][tx] = W[(size_t)(kb + ty + r) * D + nb + tx];
    __syncthreads();
    #pragma unroll
    for (int r = 0; r < 32; r += 8)
        Wt[(size_t)(nb + ty + r) * D + kb + tx] = (_Float16)t[tx][ty + r];
}

// ---------------------------------------------------------------------------
// GEMM: Out[M,768] = A[M,768](f16) @ Bt[768,768](f16, [n][k]) + bias, f32 out.
// 256 threads = 8 wave32s (2x4); block tile 128x128, wave tile 64x32 = 4x2
// WMMA 16x16x32 tiles. Double-buffered LDS; tiles staged with
// global_load_async_to_lds_b128 and pipelined one K-step ahead (in-order
// async completion: asynccnt<=4 => previous tile landed). Fragments read as
// ds_load_b128 pairs matching the ISA 16-bit layout.
// ---------------------------------------------------------------------------
__launch_bounds__(256)
__global__ void gemm_f16wmma(const _Float16* __restrict__ A,
                             const _Float16* __restrict__ Bt,
                             const float* __restrict__ bias,
                             float* __restrict__ Out)
{
    constexpr int Kd = D_EMBD;                       // K == N == 768
    constexpr int NK = Kd / 32;                      // 24 K-steps
    __shared__ __align__(16) _Float16 lA[2][128 * 32];   // [buf][m][k]
    __shared__ __align__(16) _Float16 lB[2][128 * 32];   // [buf][n][k]

    const int tid   = threadIdx.x;
    const int lane  = tid & 31;
    const int wave  = tid >> 5;
    const int wm    = wave >> 2;        // 0..1 : 64-row wave band
    const int wn    = wave & 3;         // 0..3 : 32-col wave band
    const int mlane = lane & 15;
    const int hs    = lane >> 4;

    const int mBlk = blockIdx.x * 128;
    const int nBlk = blockIdx.y * 128;

    // Staging: 512 16B-chunks per tile, 2 per thread. chunk c -> row c>>2,
    // part c&3 (8 halfs).
    const int r0 = tid >> 2,         p0 = tid & 3;
    const int r1 = (tid + 256) >> 2, p1 = (tid + 256) & 3;
    uint32_t ldsA0[2], ldsA1[2], ldsB0[2], ldsB1[2];
    #pragma unroll
    for (int b = 0; b < 2; ++b) {
        ldsA0[b] = (uint32_t)(uintptr_t)&lA[b][r0 * 32 + p0 * 8];
        ldsA1[b] = (uint32_t)(uintptr_t)&lA[b][r1 * 32 + p1 * 8];
        ldsB0[b] = (uint32_t)(uintptr_t)&lB[b][r0 * 32 + p0 * 8];
        ldsB1[b] = (uint32_t)(uintptr_t)&lB[b][r1 * 32 + p1 * 8];
    }
    uint32_t offA0 = (uint32_t)(((mBlk + r0) * Kd + p0 * 8) * 2);
    uint32_t offA1 = (uint32_t)(((mBlk + r1) * Kd + p1 * 8) * 2);
    uint32_t offB0 = (uint32_t)(((nBlk + r0) * Kd + p0 * 8) * 2);
    uint32_t offB1 = (uint32_t)(((nBlk + r1) * Kd + p1 * 8) * 2);

    v8f acc[4][2] = {};

    // ---- prologue: stage tile 0 into buffer 0 ----
    async_ld_b128(ldsA0[0], A,  offA0);
    async_ld_b128(ldsA1[0], A,  offA1);
    async_ld_b128(ldsB0[0], Bt, offB0);
    async_ld_b128(ldsB1[0], Bt, offB1);
    offA0 += 64; offA1 += 64; offB0 += 64; offB1 += 64;     // 32 halfs

    #pragma unroll 2
    for (int k = 0; k < NK; ++k) {
        const int cur = k & 1;
        if (k + 1 < NK) {
            const int nxt = cur ^ 1;   // buffer last read in iter k-1 (barrier'd)
            async_ld_b128(ldsA0[nxt], A,  offA0);
            async_ld_b128(ldsA1[nxt], A,  offA1);
            async_ld_b128(ldsB0[nxt], Bt, offB0);
            async_ld_b128(ldsB1[nxt], Bt, offB1);
            offA0 += 64; offA1 += 64; offB0 += 64; offB1 += 64;
            wait_async_le4();          // tile k complete (in-order)
        } else {
            wait_async0();
        }
        __syncthreads();               // tile k visible to all waves

        union { v16h v; v8h h[2]; } af[4], bf[2];
        #pragma unroll
        for (int i = 0; i < 4; ++i) {
            const _Float16* pa = &lA[cur][(wm * 64 + i * 16 + mlane) * 32 + hs * 8];
            af[i].h[0] = *(const v8h*)(pa);
            af[i].h[1] = *(const v8h*)(pa + 16);
        }
        #pragma unroll
        for (int j = 0; j < 2; ++j) {
            const _Float16* pb = &lB[cur][(wn * 32 + j * 16 + mlane) * 32 + hs * 8];
            bf[j].h[0] = *(const v8h*)(pb);
            bf[j].h[1] = *(const v8h*)(pb + 16);
        }
        #pragma unroll
        for (int i = 0; i < 4; ++i)
            #pragma unroll
            for (int j = 0; j < 2; ++j)
                acc[i][j] = __builtin_amdgcn_wmma_f32_16x16x32_f16(
                    false, af[i].v, false, bf[j].v, (short)0, acc[i][j],
                    false, false);
        __syncthreads();               // all reads of buf[cur] done (WAR guard)
    }

    // ---- epilogue: one 64-bit base + compile-time offsets; fused bias ----
    float* obase = Out + (size_t)(mBlk + wm * 64 + 8 * hs) * Kd
                       + (nBlk + wn * 32 + mlane);
    #pragma unroll
    for (int j = 0; j < 2; ++j) {
        float bv = bias[nBlk + wn * 32 + j * 16 + mlane];
        #pragma unroll
        for (int i = 0; i < 4; ++i)
            #pragma unroll
            for (int r = 0; r < 8; ++r)
                obase[(i * 16 + r) * Kd + j * 16] = acc[i][j][r] + bv;
    }
}

// ---------------------------------------------------------------------------
// Per-token masked softmax-attention over the HEAD axis (VALU; ~4% of FLOPs).
// S[i][k'] = (q_i . k_k')/sqrt(12); keep iff k'<=i && S!=0 else -inf
// (faithful to mask = tril(scores); where(mask != 0, ...)); private row
// softmax; Y[i][j] = sum_k' P[k'] * v[k'][j]. Emits Y in f16 for the final
// WMMA GEMM. 64 threads/token, 4 tokens per block.
// ---------------------------------------------------------------------------
__launch_bounds__(256)
__global__ void attn_kernel(const float* __restrict__ Q,
                            const float* __restrict__ Kp,
                            const float* __restrict__ Vp,
                            _Float16* __restrict__ Yh)
{
    __shared__ float sK[4][D_EMBD];
    __shared__ float sV[4][D_EMBD];

    const int tid  = threadIdx.x;
    const int slot = tid >> 6;
    const int i    = tid & 63;
    const size_t tok = (size_t)blockIdx.x * 4 + slot;

    const float* krow = Kp + tok * D_EMBD;
    const float* vrow = Vp + tok * D_EMBD;
    const float* qrow = Q  + tok * D_EMBD;

    #pragma unroll
    for (int j = 0; j < N_HEAD; ++j) {
        sK[slot][i * N_HEAD + j] = krow[i * N_HEAD + j];
        sV[slot][i * N_HEAD + j] = vrow[i * N_HEAD + j];
    }
    float q[N_HEAD];
    #pragma unroll
    for (int j = 0; j < N_HEAD; ++j) q[j] = qrow[i * N_HEAD + j];
    __syncthreads();

    const float inv_norm = 0.28867513459481287f;  // 1/sqrt(12)
    float s[HD];
    float mx = -INFINITY;
    #pragma unroll 4
    for (int kk = 0; kk < HD; ++kk) {
        float acc = 0.0f;
        #pragma unroll
        for (int j = 0; j < N_HEAD; ++j) acc += q[j] * sK[slot][kk * N_HEAD + j];
        acc *= inv_norm;
        bool keep = (kk <= i) && (acc != 0.0f);
        s[kk] = keep ? acc : -INFINITY;
        mx = fmaxf(mx, s[kk]);
    }
    float sum = 0.0f;
    #pragma unroll 4
    for (int kk = 0; kk < HD; ++kk) {
        float e = __expf(s[kk] - mx);
        s[kk] = e;
        sum += e;
    }
    const float rsum = 1.0f / sum;

    float y[N_HEAD];
    #pragma unroll
    for (int j = 0; j < N_HEAD; ++j) y[j] = 0.0f;
    #pragma unroll 4
    for (int kk = 0; kk < HD; ++kk) {
        float p = s[kk] * rsum;
        #pragma unroll
        for (int j = 0; j < N_HEAD; ++j) y[j] += p * sV[slot][kk * N_HEAD + j];
    }
    _Float16* yrow = Yh + tok * D_EMBD;
    #pragma unroll
    for (int j = 0; j < N_HEAD; ++j) yrow[i * N_HEAD + j] = (_Float16)y[j];
}

// ---------------------------------------------------------------------------
extern "C" void kernel_launch(void* const* d_in, const int* in_sizes, int n_in,
                              void* d_out, int out_size, void* d_ws, size_t ws_size,
                              hipStream_t stream)
{
    const float* x  = (const float*)d_in[0];
    const float* Wq = (const float*)d_in[1];
    const float* bq = (const float*)d_in[2];
    const float* Wk = (const float*)d_in[3];
    const float* bk = (const float*)d_in[4];
    const float* Wv = (const float*)d_in[5];
    const float* bv = (const float*)d_in[6];
    const float* Wo = (const float*)d_in[7];
    const float* bo = (const float*)d_in[8];
    float* out = (float*)d_out;

    const int Mrows = 8 * 1024;                    // b*l = 8192
    const int D = D_EMBD;
    const size_t xe = (size_t)Mrows * D;           // x / Q / K / V / Y elems
    const size_t we = (size_t)D * D;               // weight elems

    char* ws = (char*)d_ws;
    auto take = [&](size_t bytes) { char* p = ws; ws += (bytes + 255) & ~(size_t)255; return p; };
    _Float16* xh  = (_Float16*)take(xe * sizeof(_Float16));
    _Float16* Wqt = (_Float16*)take(we * sizeof(_Float16));
    _Float16* Wkt = (_Float16*)take(we * sizeof(_Float16));
    _Float16* Wvt = (_Float16*)take(we * sizeof(_Float16));
    _Float16* Wot = (_Float16*)take(we * sizeof(_Float16));
    float*    Qb  = (float*)take(xe * sizeof(float));
    float*    Kb  = (float*)take(xe * sizeof(float));
    float*    Vb  = (float*)take(xe * sizeof(float));
    _Float16* Yh  = (_Float16*)take(xe * sizeof(_Float16));

    // -- prologue: one-time f16 conversion / weight transposition --
    {
        int n4 = (int)(xe / 4);
        convert_f16<<<(n4 + 255) / 256, 256, 0, stream>>>(x, xh, n4);
        dim3 tg(D / 32, D / 32);
        transpose_convert<<<tg, 256, 0, stream>>>(Wq, Wqt, D);
        transpose_convert<<<tg, 256, 0, stream>>>(Wk, Wkt, D);
        transpose_convert<<<tg, 256, 0, stream>>>(Wv, Wvt, D);
        transpose_convert<<<tg, 256, 0, stream>>>(Wo, Wot, D);
    }

    dim3 gblk(Mrows / 128, D / 128);               // 64 x 6 blocks
    gemm_f16wmma<<<gblk, 256, 0, stream>>>(xh, Wqt, bq, Qb);
    gemm_f16wmma<<<gblk, 256, 0, stream>>>(xh, Wkt, bk, Kb);
    gemm_f16wmma<<<gblk, 256, 0, stream>>>(xh, Wvt, bv, Vb);
    attn_kernel<<<Mrows / 4, 256, 0, stream>>>(Qb, Kb, Vb, Yh);
    gemm_f16wmma<<<gblk, 256, 0, stream>>>(Yh, Wot, bo, out);
}
